// PNN_850403524971
// MI455X (gfx1250) — matrix operand order repaired and checked by reading, not compiled, not run on hardware
//
#include <hip/hip_runtime.h>

// ---------------------------------------------------------------------------
// PNN forward for MI455X (gfx1250, wave32): fused bf16x3-split WMMA GEMM.
//   d2 = ||x||^2 - 2 x.c^T + ||c||^2 ;  p = exp(-d2/2)
//   out[b,o] = (max_d p * (ND+1) - sum_d p) / ND
// Split fp32 -> bf16 hi + bf16 lo happens ONCE per element at LDS staging;
// the inner loop is pure ds_load_b128 + v_wmma_f32_16x16x32_bf16.
// ---------------------------------------------------------------------------

typedef __bf16 v16bf __attribute__((ext_vector_type(16)));
typedef __bf16 v8bf  __attribute__((ext_vector_type(8)));
typedef __bf16 v4bf  __attribute__((ext_vector_type(4)));
typedef float  v8f   __attribute__((ext_vector_type(8)));

#define B_    4096
#define IN_   1024
#define OUT_  512
#define ND_   8
#define OD_   4096   // OUT_ * ND_

#define TILE  128    // M and N tile per workgroup
#define KC    32     // K chunk staged in LDS (== WMMA K)
#define LSTRB 40     // LDS row stride in bf16 elements (32 + 8 pad, 16B-aligned rows)

__device__ __forceinline__ v16bf cat8(v8bf a, v8bf b) {
    return __builtin_shufflevector(a, b, 0,1,2,3,4,5,6,7,8,9,10,11,12,13,14,15);
}

// split one float4 into bf16 hi (RTNE) + bf16 residual lo, store both to LDS
__device__ __forceinline__ void split_store(float4 v, __bf16* hi, __bf16* lo) {
    float f[4] = {v.x, v.y, v.z, v.w};
    v4bf h, l;
#pragma unroll
    for (int i = 0; i < 4; ++i) {
        __bf16 hb = (__bf16)f[i];
        h[i] = hb;
        l[i] = (__bf16)(f[i] - (float)hb);
    }
    *(v4bf*)hi = h;
    *(v4bf*)lo = l;
}

// --------------------------------------------------------------------------
// Kernel 1: squared row norms of x (rows 0..B-1) and c (rows B..B+OD-1).
// One wave per row of 1024 floats.
// --------------------------------------------------------------------------
__global__ __launch_bounds__(256) void pnn_sqnorm(const float* __restrict__ x,
                                                  const float* __restrict__ c,
                                                  float* __restrict__ x2,
                                                  float* __restrict__ c2) {
    const int wid  = blockIdx.x * 8 + (threadIdx.x >> 5);
    const int lane = threadIdx.x & 31;
    const float* src = (wid < B_) ? (x + (size_t)wid * IN_)
                                  : (c + (size_t)(wid - B_) * IN_);
    float s = 0.f;
#pragma unroll
    for (int i = 0; i < IN_ / 128; ++i) {
        float4 v = *(const float4*)(src + i * 128 + lane * 4);
        s += v.x * v.x + v.y * v.y + v.z * v.z + v.w * v.w;
    }
#pragma unroll
    for (int m = 16; m >= 1; m >>= 1) s += __shfl_xor(s, m, 32);
    if (lane == 0) {
        if (wid < B_) x2[wid] = s;
        else          c2[wid - B_] = s;
    }
}

// --------------------------------------------------------------------------
// Kernel 2: fused tiled GEMM (bf16x3 WMMA) + PNN epilogue.
// Block: 256 threads = 8 waves. Tile: 128(M) x 128(N). Wave w owns rows
// [16w,16w+16) of the tile across all 128 N columns -> 8 v8f accumulators.
// --------------------------------------------------------------------------
__global__ __launch_bounds__(256) void pnn_main(const float* __restrict__ x,
                                                const float* __restrict__ c,
                                                const float* __restrict__ x2,
                                                const float* __restrict__ c2,
                                                float* __restrict__ out) {
    __shared__ __bf16 lXh[TILE * LSTRB];
    __shared__ __bf16 lXl[TILE * LSTRB];
    __shared__ __bf16 lCh[TILE * LSTRB];
    __shared__ __bf16 lCl[TILE * LSTRB];

    const int tid   = threadIdx.x;
    const int wave  = tid >> 5;
    const int lane  = tid & 31;
    const int tileM = blockIdx.y * TILE;
    const int tileN = blockIdx.x * TILE;

    v8f acc[8];
#pragma unroll
    for (int n = 0; n < 8; ++n) acc[n] = (v8f){0.f,0.f,0.f,0.f,0.f,0.f,0.f,0.f};

    const int mrow = 16 * wave + (lane & 15);   // A-fragment LDS row
    const int kbA  = (lane >> 4) * 8;           // A layout: runs [kb,kb+8),[kb+16,kb+24)
    const int kbB  = (lane >> 4) * 8;           // B layout: runs [kb,kb+8),[kb+16,kb+24)

    // staging indices for this thread: 4 float4 segments per matrix per chunk
    int srow[4], sseg[4];
#pragma unroll
    for (int i = 0; i < 4; ++i) {
        int idx = tid + i * 256;
        srow[i] = idx >> 3;
        sseg[i] = idx & 7;
    }

    float4 nx[4], nc[4];
    // ---- prologue: load + convert + store chunk 0 --------------------------
#pragma unroll
    for (int i = 0; i < 4; ++i) {
        nx[i] = *(const float4*)(x + (size_t)(tileM + srow[i]) * IN_ + sseg[i] * 4);
        nc[i] = *(const float4*)(c + (size_t)(tileN + srow[i]) * IN_ + sseg[i] * 4);
    }
#pragma unroll
    for (int i = 0; i < 4; ++i) {
        int o = srow[i] * LSTRB + sseg[i] * 4;
        split_store(nx[i], &lXh[o], &lXl[o]);
        split_store(nc[i], &lCh[o], &lCl[o]);
    }
    __syncthreads();

    for (int kc = 0; kc < IN_; kc += KC) {
        const bool more = (kc + KC) < IN_;
        // issue next chunk's global loads; latency hides under the WMMAs below
        if (more) {
#pragma unroll
            for (int i = 0; i < 4; ++i) {
                nx[i] = *(const float4*)(x + (size_t)(tileM + srow[i]) * IN_ + kc + KC + sseg[i] * 4);
                nc[i] = *(const float4*)(c + (size_t)(tileN + srow[i]) * IN_ + kc + KC + sseg[i] * 4);
            }
        }

        // ---- compute: pure ds_load_b128 + wmma ----------------------------
        const int abase = mrow * LSTRB + kbA;
        v16bf ah = cat8(*(const v8bf*)&lXh[abase], *(const v8bf*)&lXh[abase + 16]);
        v16bf al = cat8(*(const v8bf*)&lXl[abase], *(const v8bf*)&lXl[abase + 16]);

#pragma unroll
        for (int n = 0; n < 8; ++n) {
            const int bbase = (16 * n + (lane & 15)) * LSTRB + kbB;
            v16bf bh = cat8(*(const v8bf*)&lCh[bbase], *(const v8bf*)&lCh[bbase + 16]);
            v16bf bl = cat8(*(const v8bf*)&lCl[bbase], *(const v8bf*)&lCl[bbase + 16]);

            acc[n] = __builtin_amdgcn_wmma_f32_16x16x32_bf16(
                false, ah, false, bh, (short)0, acc[n], false, false);
            acc[n] = __builtin_amdgcn_wmma_f32_16x16x32_bf16(
                false, ah, false, bl, (short)0, acc[n], false, false);
            acc[n] = __builtin_amdgcn_wmma_f32_16x16x32_bf16(
                false, al, false, bh, (short)0, acc[n], false, false);
        }
        __syncthreads();               // all waves done reading this chunk

        if (more) {
#pragma unroll
            for (int i = 0; i < 4; ++i) {
                int o = srow[i] * LSTRB + sseg[i] * 4;
                split_store(nx[i], &lXh[o], &lXl[o]);
                split_store(nc[i], &lCh[o], &lCl[o]);
            }
            __syncthreads();           // next chunk visible to all waves
        }
    }

    // ---- epilogue: d2 -> exp -> per-8-column (ND) max/sum reduction --------
    // C/D layout: element g of lane L is (M = g + 8*(L>>4), N = L&15)
    const int half   = lane >> 4;
    const int nlocal = lane & 15;
    float xs[8];
#pragma unroll
    for (int g = 0; g < 8; ++g)
        xs[g] = x2[tileM + 16 * wave + 8 * half + g];

#pragma unroll
    for (int n = 0; n < 8; ++n) {
        const int col = tileN + 16 * n + nlocal;
        const float cs = c2[col];
#pragma unroll
        for (int g = 0; g < 8; ++g) {
            float d2 = xs[g] - 2.0f * acc[n][g] + cs;
            float p  = __expf(-0.5f * d2);
            float pmax = p, psum = p;
            // xor-butterfly within aligned groups of 8 lanes == one ND group
#pragma unroll
            for (int m = 1; m <= 4; m <<= 1) {
                pmax = fmaxf(pmax, __shfl_xor(pmax, m, 32));
                psum += __shfl_xor(psum, m, 32);
            }
            if ((lane & 7) == 0) {
                int orow = tileM + 16 * wave + 8 * half + g;
                int ocol = ((tileN + 16 * n) >> 3) + ((lane >> 3) & 1);
                out[(size_t)orow * OUT_ + ocol] =
                    (pmax * (float)(ND_ + 1) - psum) * (1.0f / (float)ND_);
            }
        }
    }
}

extern "C" void kernel_launch(void* const* d_in, const int* in_sizes, int n_in,
                              void* d_out, int out_size, void* d_ws, size_t ws_size,
                              hipStream_t stream) {
    (void)in_sizes; (void)n_in; (void)out_size; (void)ws_size;
    const float* x       = (const float*)d_in[0];
    const float* centers = (const float*)d_in[1];
    float*       out     = (float*)d_out;
    float*       x2      = (float*)d_ws;          // B_ floats
    float*       c2      = x2 + B_;               // OD_ floats (32 KB total)

    pnn_sqnorm<<<dim3((B_ + OD_) / 8), dim3(256), 0, stream>>>(x, centers, x2, c2);
    pnn_main<<<dim3(OD_ / TILE, B_ / TILE), dim3(256), 0, stream>>>(x, centers, x2, c2, out);
}
